// STTM_Single_37555194036286
// MI455X (gfx1250) — compile-verified, matrix operand order checked by reading, not run on hardware
//
#include <hip/hip_runtime.h>
#include <hip/hip_bf16.h>
#include <stdint.h>

// ---------------- problem dims ----------------
#define DIMC   512
#define BATCH  2
#define HH     96
#define WWID   320
#define HWSZ   (HH * WWID)        // 30720
#define NPIX   (BATCH * HWSZ)     // 61440
#define NHEADS 8
#define DHEAD  64
#define OUTC   256
#define ATTNSCALE 0.125f          // 64^-0.5
#define BN_EPS 1e-5f

// ---------------- types ----------------
typedef __attribute__((ext_vector_type(16))) __bf16 v16bf;
typedef __attribute__((ext_vector_type(8)))  float  v8f;
typedef __attribute__((ext_vector_type(4)))  unsigned int u32x4;
typedef __attribute__((ext_vector_type(8)))  int i32x8;
typedef __attribute__((ext_vector_type(4)))  int i32x4;

union FragU { v16bf v; float4 q[2]; };

__device__ __forceinline__ v8f vzero() {
  v8f z = {0.f, 0.f, 0.f, 0.f, 0.f, 0.f, 0.f, 0.f};
  return z;
}

// LDS byte address of a shared-memory pointer (LDS offset = addr[31:0])
__device__ __forceinline__ unsigned lds_addr_of(const void* p) {
  return (unsigned)(unsigned long long)p;
}

// A-fragment (and TN B-fragment) loader: row-major [16 x 32] bf16 tile in LDS.
// lane 0-15: M=lane, K=0..7 & 16..23 ; lane 16-31: M=lane-16, K=8..15 & 24..31
__device__ __forceinline__ v16bf lds_frag(const __bf16* base, int strideElems, int lane) {
  int row = lane & 15;
  int ko  = (lane >> 4) << 3;                 // 0 or 8 elements
  const __bf16* r = base + row * strideElems + ko;
  FragU f;
  f.q[0] = *(const float4*)(r);               // K = ko .. ko+7
  f.q[1] = *(const float4*)(r + 16);          // K = ko+16 .. ko+23
  return f.v;
}

__device__ __forceinline__ v8f wmma_bf16(v16bf a, v16bf b, v8f c) {
  return __builtin_amdgcn_wmma_f32_16x16x32_bf16(false, a, false, b, (short)0, c,
                                                 false, false);
}

// C-fragment bf16 scatter: lane 0-15: N=lane, M=0..7 ; lane 16-31: N=lane-16, M=8..15
__device__ __forceinline__ void scatter_bf16(__bf16* base, int strideElems, int lane, v8f c) {
  int col = lane & 15;
  int m0  = (lane >> 4) << 3;
#pragma unroll
  for (int r = 0; r < 8; ++r)
    base[(size_t)(m0 + r) * strideElems + col] = (__bf16)c[r];
}

// ---------------- Tensor Data Mover (async 2D tile -> LDS) ----------------
#if defined(__has_builtin)
#if __has_builtin(__builtin_amdgcn_tensor_load_to_lds)
#define HAVE_TDM 1
#endif
#endif

#ifdef HAVE_TDM
// 2D tile of bf16: n_rows x n_cols elements, global row stride g_stride elems,
// landed at LDS byte address lds_off with optional row padding.
__device__ __forceinline__ void tdm_load_2d_bf16(unsigned lds_off, const void* gaddr,
                                                 unsigned n_rows, unsigned n_cols,
                                                 unsigned g_stride,
                                                 unsigned pad_int_code, unsigned pad_amt_code,
                                                 bool pad) {
  unsigned long long ga = (unsigned long long)gaddr;
  u32x4 g0;
  g0[0] = 1u;                                            // count=1, user mode
  g0[1] = lds_off;                                       // lds_addr (bytes)
  g0[2] = (unsigned)(ga & 0xFFFFFFFFull);                // global_addr lo
  g0[3] = (unsigned)((ga >> 32) & 0x1FFFFFFull) | 0x80000000u;  // addr hi | type=2
  i32x8 g1;
  unsigned d0 = (1u << 16);                              // data_size = 2 bytes
  if (pad) d0 |= (1u << 20) | (pad_int_code << 22) | (pad_amt_code << 25);
  g1[0] = (int)d0;
  g1[1] = (int)((n_cols & 0xFFFFu) << 16);               // tensor_dim0[15:0]
  g1[2] = (int)(((n_cols >> 16) & 0xFFFFu) | ((n_rows & 0xFFFFu) << 16)); // dim0 hi | dim1 lo
  g1[3] = (int)(((n_rows >> 16) & 0xFFFFu) | ((n_cols & 0xFFFFu) << 16)); // dim1 hi | tile_dim0
  g1[4] = (int)(n_rows & 0xFFFFu);                       // tile_dim1 (tile_dim2 = 0)
  g1[5] = (int)g_stride;                                 // tensor_dim0_stride lo32
  g1[6] = 0;
  g1[7] = 0;
  i32x4 z4 = {0, 0, 0, 0};
  i32x8 z8 = {0, 0, 0, 0, 0, 0, 0, 0};
  __builtin_amdgcn_tensor_load_to_lds(g0, g1, z4, z4, z8, 0);
}
#endif

// ======================================================================
// K0a: fp32 [b,c,h,w] -> bf16 pixel-major Xt[p][512], LDS-tiled transpose
// ======================================================================
__global__ void k_convT(const float* __restrict__ src, __bf16* __restrict__ dst) {
  __shared__ float tile[32][33];
  int p0 = blockIdx.x * 32;
  int c0 = blockIdx.y * 32;
  int b  = p0 / HWSZ;
  int hw0 = p0 - b * HWSZ;
  const float* s = src + (size_t)b * DIMC * HWSZ + (size_t)c0 * HWSZ + hw0;
#pragma unroll
  for (int i = 0; i < 4; ++i) {
    int c = threadIdx.y + i * 8;
    tile[c][threadIdx.x] = s[(size_t)c * HWSZ + threadIdx.x];
  }
  __syncthreads();
#pragma unroll
  for (int i = 0; i < 4; ++i) {
    int pr = threadIdx.y + i * 8;
    dst[(size_t)(p0 + pr) * DIMC + c0 + threadIdx.x] = (__bf16)tile[threadIdx.x][pr];
  }
}

// K0b: flat fp32 -> bf16
__global__ void k_cvt(const float* __restrict__ src, __bf16* __restrict__ dst, int n) {
  int i = blockIdx.x * blockDim.x + threadIdx.x;
  if (i < n) dst[i] = (__bf16)src[i];
}

// K0c: fold BN running stats into scale/shift
__global__ void k_bnprep(const float* __restrict__ g, const float* __restrict__ be,
                         const float* __restrict__ mu, const float* __restrict__ va,
                         float* __restrict__ sc, float* __restrict__ sh) {
  int i = threadIdx.x;
  if (i < OUTC) {
    float s = g[i] * rsqrtf(va[i] + BN_EPS);
    sc[i] = s;
    sh[i] = be[i] - mu[i] * s;
  }
}

// ======================================================================
// Double-buffered TN-GEMM body: WG tile 128x128, wave tile 64x64,
// 16 WMMA per k-step. Staging via TDM (no VGPR staging -> no spills).
// ======================================================================
#define GSTRIDE 40   // LDS row stride (elems) for a 32-wide k-chunk
// GEMM tile rows of 32 bf16 = 16 dwords -> pad_interval code 3;
// pad 8 elems = 4 dwords -> pad_amount code 2  (=> 40-elem stride)

struct GemmAcc { v8f a[4][4]; };

__device__ __forceinline__ void gemm_step(GemmAcc& acc,
                                          __bf16 (*As)[GSTRIDE], __bf16 (*Bs)[GSTRIDE],
                                          int wm, int wn, int lane) {
  v16bf af[4];
#pragma unroll
  for (int ia = 0; ia < 4; ++ia)
    af[ia] = lds_frag(&As[wm * 64 + ia * 16][0], GSTRIDE, lane);
#pragma unroll
  for (int ib = 0; ib < 4; ++ib) {
    v16bf bf_ = lds_frag(&Bs[wn * 64 + ib * 16][0], GSTRIDE, lane);
#pragma unroll
    for (int ia = 0; ia < 4; ++ia)
      acc.a[ia][ib] = wmma_bf16(af[ia], bf_, acc.a[ia][ib]);
  }
}

#ifndef HAVE_TDM
// Fallback: plain cooperative copy of one 128x32 chunk into LDS
__device__ __forceinline__ void stage_copy(__bf16 (*buf)[GSTRIDE], int tid,
                                           const __bf16* srcRow) {
  *(float4*)&buf[tid][0]  = *(const float4*)(srcRow);
  *(float4*)&buf[tid][8]  = *(const float4*)(srcRow + 8);
  *(float4*)&buf[tid][16] = *(const float4*)(srcRow + 16);
  *(float4*)&buf[tid][24] = *(const float4*)(srcRow + 24);
}
#endif

// ======================================================================
// K1: QKV projection.  C[p,o] = Xt[p,:] . W[o,:]   K=512
// grid (480, 4, 3)  block 128
// ======================================================================
__global__ __launch_bounds__(128, 1) void k_proj(
    const __bf16* __restrict__ XtL, const __bf16* __restrict__ XtR,
    const __bf16* __restrict__ Wq, const __bf16* __restrict__ Wk,
    const __bf16* __restrict__ Wv,
    __bf16* __restrict__ Q, __bf16* __restrict__ K, __bf16* __restrict__ V) {
  __shared__ __bf16 As[2][128][GSTRIDE];
  __shared__ __bf16 Bs[2][128][GSTRIDE];
  int which = blockIdx.z;
  const __bf16* X  = (which == 0) ? XtL : XtR;
  const __bf16* Wm = (which == 0) ? Wq : (which == 1 ? Wk : Wv);
  __bf16* Out      = (which == 0) ? Q  : (which == 1 ? K  : V);

  int pt = blockIdx.x * 128;
  int nt = blockIdx.y * 128;
  int tid = threadIdx.x, lane = tid & 31, wave = tid >> 5;
  int wm = wave >> 1, wn = wave & 1;

  GemmAcc acc;
#pragma unroll
  for (int i = 0; i < 4; ++i)
#pragma unroll
    for (int j = 0; j < 4; ++j) acc.a[i][j] = vzero();

  const __bf16* abase = X  + (size_t)pt * DIMC;
  const __bf16* bbase = Wm + (size_t)nt * DIMC;

#ifdef HAVE_TDM
  unsigned la0 = lds_addr_of(&As[0][0][0]), la1 = lds_addr_of(&As[1][0][0]);
  unsigned lb0 = lds_addr_of(&Bs[0][0][0]), lb1 = lds_addr_of(&Bs[1][0][0]);
  if (tid < 32) {
    tdm_load_2d_bf16(la0, abase, 128, 32, DIMC, 3, 2, true);
    tdm_load_2d_bf16(lb0, bbase, 128, 32, DIMC, 3, 2, true);
  }
  for (int kc = 0; kc < DIMC; kc += 32) {
    int buf = (kc >> 5) & 1;
    if (tid < 32) {
      if (kc + 32 < DIMC) {
        tdm_load_2d_bf16(buf ? la0 : la1, abase + kc + 32, 128, 32, DIMC, 3, 2, true);
        tdm_load_2d_bf16(buf ? lb0 : lb1, bbase + kc + 32, 128, 32, DIMC, 3, 2, true);
        __builtin_amdgcn_s_wait_tensorcnt(2);   // current chunk landed (in-order)
      } else {
        __builtin_amdgcn_s_wait_tensorcnt(0);
      }
    }
    __syncthreads();
    gemm_step(acc, As[buf], Bs[buf], wm, wn, lane);
    __syncthreads();
  }
#else
  for (int kc = 0; kc < DIMC; kc += 32) {
    stage_copy(As[0], tid, abase + (size_t)tid * DIMC + kc);
    stage_copy(Bs[0], tid, bbase + (size_t)tid * DIMC + kc);
    __syncthreads();
    gemm_step(acc, As[0], Bs[0], wm, wn, lane);
    __syncthreads();
  }
#endif
#pragma unroll
  for (int ia = 0; ia < 4; ++ia)
#pragma unroll
    for (int ib = 0; ib < 4; ++ib)
      scatter_bf16(&Out[(size_t)(pt + wm * 64 + ia * 16) * DIMC + nt + wn * 64 + ib * 16],
                   DIMC, lane, acc.a[ia][ib]);
}

// ======================================================================
// K2: cross attention per (b*H row, head). 1 WG = 4 waves, 1536 WGs.
// LDS: Qs[320][72] Ks[320][72] Vt[64][328] P[4][16][336]  (~173 KB)
// ======================================================================
#define QK_STRIDE 72
#define QS_BYTES  (WWID * QK_STRIDE * 2)      // 46080
#define VT_STRIDE 328
#define VT_BYTES  (DHEAD * VT_STRIDE * 2)     // 41984
#define P_STRIDE  336
#define PW_BYTES  (16 * P_STRIDE * 2)         // 10752
#define ATTN_LDS  (2 * QS_BYTES + VT_BYTES + 4 * PW_BYTES)   // 177152

__global__ __launch_bounds__(128, 1) void k_attn(
    const __bf16* __restrict__ Qg, const __bf16* __restrict__ Kg,
    const __bf16* __restrict__ Vg, __bf16* __restrict__ Og) {
  extern __shared__ __align__(16) char smem[];
  __bf16* Qs = (__bf16*)smem;
  __bf16* Ks = (__bf16*)(smem + QS_BYTES);
  __bf16* Vt = (__bf16*)(smem + 2 * QS_BYTES);
  __bf16* Ps = (__bf16*)(smem + 2 * QS_BYTES + VT_BYTES);

  int bh   = blockIdx.x >> 3;
  int head = blockIdx.x & 7;
  size_t p0 = (size_t)bh * WWID;
  const __bf16* qbase = Qg + p0 * DIMC + head * DHEAD;
  const __bf16* kbase = Kg + p0 * DIMC + head * DHEAD;
  const __bf16* vbase = Vg + p0 * DIMC + head * DHEAD;

  int tid = threadIdx.x, lane = tid & 31, wave = tid >> 5;

#ifdef HAVE_TDM
  if (tid < 32) {
    // row = 64 bf16 = 32 dwords -> pad_interval code 4 ; pad 4 dwords -> code 2
    tdm_load_2d_bf16(lds_addr_of(Qs), qbase, WWID, DHEAD, DIMC, 4, 2, true);
    tdm_load_2d_bf16(lds_addr_of(Ks), kbase, WWID, DHEAD, DIMC, 4, 2, true);
    __builtin_amdgcn_s_wait_tensorcnt(0);
  }
#else
  for (int i = tid; i < WWID * DHEAD; i += 128) {
    int x = i >> 6, d = i & 63;
    Qs[x * QK_STRIDE + d] = qbase[(size_t)x * DIMC + d];
    Ks[x * QK_STRIDE + d] = kbase[(size_t)x * DIMC + d];
  }
#endif
  // Stage V transposed: Vt[d][x]  (so P.V is a TN GEMM with plain frags)
  for (int i = tid; i < WWID * DHEAD; i += 128) {
    int x = i >> 6, d = i & 63;
    Vt[d * VT_STRIDE + x] = vbase[(size_t)x * DIMC + d];
  }
  __syncthreads();

  __bf16* Pw = Ps + (size_t)wave * (16 * P_STRIDE);
  int prow0 = (lane >> 4) << 3;
  int pcol  = lane & 15;

  for (int mt = wave; mt < WWID / 16; mt += 4) {
    v16bf qa0 = lds_frag(Qs + mt * 16 * QK_STRIDE,      QK_STRIDE, lane);
    v16bf qa1 = lds_frag(Qs + mt * 16 * QK_STRIDE + 32, QK_STRIDE, lane);

    v8f s[20];
#pragma unroll
    for (int ntl = 0; ntl < 20; ++ntl) {
      v16bf kb0 = lds_frag(Ks + ntl * 16 * QK_STRIDE,      QK_STRIDE, lane);
      v16bf kb1 = lds_frag(Ks + ntl * 16 * QK_STRIDE + 32, QK_STRIDE, lane);
      s[ntl] = wmma_bf16(qa0, kb0, vzero());
      s[ntl] = wmma_bf16(qa1, kb1, s[ntl]);
    }

    // row-wise softmax (rows live across the 16-lane halves)
    float mx[8];
#pragma unroll
    for (int r = 0; r < 8; ++r) mx[r] = -1e30f;
#pragma unroll
    for (int ntl = 0; ntl < 20; ++ntl)
#pragma unroll
      for (int r = 0; r < 8; ++r) mx[r] = fmaxf(mx[r], s[ntl][r]);
#pragma unroll
    for (int d = 1; d < 16; d <<= 1)
#pragma unroll
      for (int r = 0; r < 8; ++r) mx[r] = fmaxf(mx[r], __shfl_xor(mx[r], d, 32));

    float sm[8];
#pragma unroll
    for (int r = 0; r < 8; ++r) sm[r] = 0.f;
#pragma unroll
    for (int ntl = 0; ntl < 20; ++ntl) {
#pragma unroll
      for (int r = 0; r < 8; ++r) {
        float e = __expf((s[ntl][r] - mx[r]) * ATTNSCALE);
        sm[r] += e;
        Pw[(size_t)(prow0 + r) * P_STRIDE + ntl * 16 + pcol] = (__bf16)e;
      }
    }
#pragma unroll
    for (int d = 1; d < 16; d <<= 1)
#pragma unroll
      for (int r = 0; r < 8; ++r) sm[r] += __shfl_xor(sm[r], d, 32);
    float rv[8];
#pragma unroll
    for (int r = 0; r < 8; ++r) rv[r] = 1.f / sm[r];

    // O = P . V
    v8f o[4];
#pragma unroll
    for (int dt = 0; dt < 4; ++dt) o[dt] = vzero();
#pragma unroll
    for (int kt = 0; kt < 10; ++kt) {
      v16bf pa = lds_frag(Pw + kt * 32, P_STRIDE, lane);
#pragma unroll
      for (int dt = 0; dt < 4; ++dt) {
        v16bf vb = lds_frag(Vt + dt * 16 * VT_STRIDE + kt * 32, VT_STRIDE, lane);
        o[dt] = wmma_bf16(pa, vb, o[dt]);
      }
    }
#pragma unroll
    for (int dt = 0; dt < 4; ++dt) {
#pragma unroll
      for (int r = 0; r < 8; ++r) o[dt][r] *= rv[r];
      scatter_bf16(Og + (p0 + mt * 16) * DIMC + head * DHEAD + dt * 16, DIMC, lane, o[dt]);
    }
  }
}

// ======================================================================
// K3: y = LeakyReLU(BN(w1 . concat(left, attn_out)))   K=1024, N=256
// grid (480, 2), block 128
// ======================================================================
__global__ __launch_bounds__(128, 1) void k_out1(
    const __bf16* __restrict__ XtL, const __bf16* __restrict__ O,
    const __bf16* __restrict__ W1, const float* __restrict__ bnsc,
    const float* __restrict__ bnsh, __bf16* __restrict__ Y) {
  __shared__ __bf16 As[2][128][GSTRIDE];
  __shared__ __bf16 Bs[2][128][GSTRIDE];
  int pt = blockIdx.x * 128;
  int nt = blockIdx.y * 128;
  int tid = threadIdx.x, lane = tid & 31, wave = tid >> 5;
  int wm = wave >> 1, wn = wave & 1;

  GemmAcc acc;
#pragma unroll
  for (int i = 0; i < 4; ++i)
#pragma unroll
    for (int j = 0; j < 4; ++j) acc.a[i][j] = vzero();

  const __bf16* abaseL = XtL + (size_t)pt * DIMC;
  const __bf16* abaseO = O   + (size_t)pt * DIMC;
  const __bf16* bbase  = W1  + (size_t)nt * 1024;

#ifdef HAVE_TDM
  unsigned la0 = lds_addr_of(&As[0][0][0]), la1 = lds_addr_of(&As[1][0][0]);
  unsigned lb0 = lds_addr_of(&Bs[0][0][0]), lb1 = lds_addr_of(&Bs[1][0][0]);
  if (tid < 32) {
    tdm_load_2d_bf16(la0, abaseL, 128, 32, DIMC, 3, 2, true);
    tdm_load_2d_bf16(lb0, bbase,  128, 32, 1024, 3, 2, true);
  }
  for (int kc = 0; kc < 1024; kc += 32) {
    int buf = (kc >> 5) & 1;
    if (tid < 32) {
      int kn = kc + 32;
      if (kn < 1024) {
        const __bf16* an = (kn < 512) ? (abaseL + kn) : (abaseO + kn - 512);
        tdm_load_2d_bf16(buf ? la0 : la1, an,         128, 32, DIMC, 3, 2, true);
        tdm_load_2d_bf16(buf ? lb0 : lb1, bbase + kn, 128, 32, 1024, 3, 2, true);
        __builtin_amdgcn_s_wait_tensorcnt(2);
      } else {
        __builtin_amdgcn_s_wait_tensorcnt(0);
      }
    }
    __syncthreads();
    gemm_step(acc, As[buf], Bs[buf], wm, wn, lane);
    __syncthreads();
  }
#else
  for (int kc = 0; kc < 1024; kc += 32) {
    const __bf16* an = (kc < 512) ? (abaseL + (size_t)tid * DIMC + kc)
                                  : (abaseO + (size_t)tid * DIMC + kc - 512);
    stage_copy(As[0], tid, an);
    stage_copy(Bs[0], tid, bbase + (size_t)tid * 1024 + kc);
    __syncthreads();
    gemm_step(acc, As[0], Bs[0], wm, wn, lane);
    __syncthreads();
  }
#endif

  int m0 = (lane >> 4) << 3;
#pragma unroll
  for (int ia = 0; ia < 4; ++ia)
#pragma unroll
    for (int ib = 0; ib < 4; ++ib) {
      int oc = nt + wn * 64 + ib * 16 + (lane & 15);
      float sc = bnsc[oc], sh = bnsh[oc];
      int pr = pt + wm * 64 + ia * 16 + m0;
#pragma unroll
      for (int r = 0; r < 8; ++r) {
        float v = acc.a[ia][ib][r] * sc + sh;
        v = (v >= 0.f) ? v : 0.2f * v;
        Y[(size_t)(pr + r) * OUTC + oc] = (__bf16)v;
      }
    }
}

// ======================================================================
// K4: out = w2 . y   K=256, N=256, fp32 output in [b,o,h,w]
// grid (480, 2), block 128
// ======================================================================
__global__ __launch_bounds__(128, 1) void k_out2(
    const __bf16* __restrict__ Y, const __bf16* __restrict__ W2,
    float* __restrict__ Outp) {
  __shared__ __bf16 As[2][128][GSTRIDE];
  __shared__ __bf16 Bs[2][128][GSTRIDE];
  int pt = blockIdx.x * 128;
  int nt = blockIdx.y * 128;
  int tid = threadIdx.x, lane = tid & 31, wave = tid >> 5;
  int wm = wave >> 1, wn = wave & 1;

  GemmAcc acc;
#pragma unroll
  for (int i = 0; i < 4; ++i)
#pragma unroll
    for (int j = 0; j < 4; ++j) acc.a[i][j] = vzero();

  const __bf16* abase = Y  + (size_t)pt * OUTC;
  const __bf16* bbase = W2 + (size_t)nt * OUTC;

#ifdef HAVE_TDM
  unsigned la0 = lds_addr_of(&As[0][0][0]), la1 = lds_addr_of(&As[1][0][0]);
  unsigned lb0 = lds_addr_of(&Bs[0][0][0]), lb1 = lds_addr_of(&Bs[1][0][0]);
  if (tid < 32) {
    tdm_load_2d_bf16(la0, abase, 128, 32, OUTC, 3, 2, true);
    tdm_load_2d_bf16(lb0, bbase, 128, 32, OUTC, 3, 2, true);
  }
  for (int kc = 0; kc < OUTC; kc += 32) {
    int buf = (kc >> 5) & 1;
    if (tid < 32) {
      if (kc + 32 < OUTC) {
        tdm_load_2d_bf16(buf ? la0 : la1, abase + kc + 32, 128, 32, OUTC, 3, 2, true);
        tdm_load_2d_bf16(buf ? lb0 : lb1, bbase + kc + 32, 128, 32, OUTC, 3, 2, true);
        __builtin_amdgcn_s_wait_tensorcnt(2);
      } else {
        __builtin_amdgcn_s_wait_tensorcnt(0);
      }
    }
    __syncthreads();
    gemm_step(acc, As[buf], Bs[buf], wm, wn, lane);
    __syncthreads();
  }
#else
  for (int kc = 0; kc < OUTC; kc += 32) {
    stage_copy(As[0], tid, abase + (size_t)tid * OUTC + kc);
    stage_copy(Bs[0], tid, bbase + (size_t)tid * OUTC + kc);
    __syncthreads();
    gemm_step(acc, As[0], Bs[0], wm, wn, lane);
    __syncthreads();
  }
#endif

  int m0 = (lane >> 4) << 3;
#pragma unroll
  for (int ia = 0; ia < 4; ++ia)
#pragma unroll
    for (int ib = 0; ib < 4; ++ib) {
      int o2 = nt + wn * 64 + ib * 16 + (lane & 15);
      int pr = pt + wm * 64 + ia * 16 + m0;    // 8-aligned; never crosses batch
      int b  = pr / HWSZ;
      int hw = pr - b * HWSZ;
      float* dst = Outp + ((size_t)b * OUTC + o2) * HWSZ + hw;
      v8f c = acc.a[ia][ib];
      float4 lo = make_float4(c[0], c[1], c[2], c[3]);
      float4 hi = make_float4(c[4], c[5], c[6], c[7]);
      *(float4*)(dst)     = lo;
      *(float4*)(dst + 4) = hi;
    }
}

// ======================================================================
extern "C" void kernel_launch(void* const* d_in, const int* in_sizes, int n_in,
                              void* d_out, int out_size, void* d_ws, size_t ws_size,
                              hipStream_t stream) {
  (void)in_sizes; (void)n_in; (void)out_size; (void)ws_size;
  const float* left  = (const float*)d_in[0];
  const float* right = (const float*)d_in[1];
  const float* wq    = (const float*)d_in[2];
  const float* wk    = (const float*)d_in[3];
  const float* wv    = (const float*)d_in[4];
  const float* w1    = (const float*)d_in[5];
  const float* bng   = (const float*)d_in[6];
  const float* bnb   = (const float*)d_in[7];
  const float* bnm   = (const float*)d_in[8];
  const float* bnv   = (const float*)d_in[9];
  const float* w2    = (const float*)d_in[10];
  float* outp = (float*)d_out;

  char* ws = (char*)d_ws;
  size_t off = 0;
  auto alloc = [&](size_t bytes) {
    char* p = ws + off;
    off += (bytes + 255) & ~(size_t)255;
    return p;
  };
  __bf16* XtL  = (__bf16*)alloc((size_t)NPIX * DIMC * 2);
  __bf16* XtR  = (__bf16*)alloc((size_t)NPIX * DIMC * 2);
  __bf16* Wqb  = (__bf16*)alloc((size_t)DIMC * DIMC * 2);
  __bf16* Wkb  = (__bf16*)alloc((size_t)DIMC * DIMC * 2);
  __bf16* Wvb  = (__bf16*)alloc((size_t)DIMC * DIMC * 2);
  __bf16* W1b  = (__bf16*)alloc((size_t)OUTC * 1024 * 2);
  __bf16* W2b  = (__bf16*)alloc((size_t)OUTC * OUTC * 2);
  float*  bnsc = (float*)alloc(OUTC * 4);
  float*  bnsh = (float*)alloc(OUTC * 4);
  __bf16* Qb   = (__bf16*)alloc((size_t)NPIX * DIMC * 2);
  __bf16* Kb   = (__bf16*)alloc((size_t)NPIX * DIMC * 2);
  __bf16* Vb   = (__bf16*)alloc((size_t)NPIX * DIMC * 2);
  __bf16* Ob   = (__bf16*)alloc((size_t)NPIX * DIMC * 2);
  __bf16* Yb   = (__bf16*)alloc((size_t)NPIX * OUTC * 2);

  dim3 bT(32, 8);
  k_convT<<<dim3(NPIX / 32, DIMC / 32), bT, 0, stream>>>(left, XtL);
  k_convT<<<dim3(NPIX / 32, DIMC / 32), bT, 0, stream>>>(right, XtR);
  k_cvt<<<(DIMC * DIMC + 255) / 256, 256, 0, stream>>>(wq, Wqb, DIMC * DIMC);
  k_cvt<<<(DIMC * DIMC + 255) / 256, 256, 0, stream>>>(wk, Wkb, DIMC * DIMC);
  k_cvt<<<(DIMC * DIMC + 255) / 256, 256, 0, stream>>>(wv, Wvb, DIMC * DIMC);
  k_cvt<<<(OUTC * 1024 + 255) / 256, 256, 0, stream>>>(w1, W1b, OUTC * 1024);
  k_cvt<<<(OUTC * OUTC + 255) / 256, 256, 0, stream>>>(w2, W2b, OUTC * OUTC);
  k_bnprep<<<1, 256, 0, stream>>>(bng, bnb, bnm, bnv, bnsc, bnsh);

  k_proj<<<dim3(NPIX / 128, DIMC / 128, 3), 128, 0, stream>>>(
      XtL, XtR, Wqb, Wkb, Wvb, Qb, Kb, Vb);

  (void)hipFuncSetAttribute((const void*)k_attn,
                            hipFuncAttributeMaxDynamicSharedMemorySize, ATTN_LDS);
  k_attn<<<BATCH * HH * NHEADS, 128, ATTN_LDS, stream>>>(Qb, Kb, Vb, Ob);

  k_out1<<<dim3(NPIX / 128, OUTC / 128), 128, 0, stream>>>(XtL, Ob, W1b, bnsc, bnsh, Yb);
  k_out2<<<dim3(NPIX / 128, OUTC / 128), 128, 0, stream>>>(Yb, W2b, outp);
}
